// AttentionWrapper_43396349559565
// MI455X (gfx1250) — compile-verified, matrix operand order checked by reading, not run on hardware
//
#include <hip/hip_runtime.h>
#include <hip/hip_bf16.h>
#include <math.h>

// Shapes
#define BB   2
#define SS   2048
#define HH   768
#define NHH  12
#define HDD  64
#define MM   (BB*SS)        // 4096

typedef __attribute__((ext_vector_type(16))) __bf16 v16bf;
typedef __attribute__((ext_vector_type(8)))  float  v8f;
typedef __attribute__((ext_vector_type(4)))  unsigned u32x4;

static __device__ __forceinline__ __bf16 f2bf(float f) {
    unsigned u = __builtin_bit_cast(unsigned, f);
    unsigned lsb = (u >> 16) & 1u;
    u += 0x7FFFu + lsb;                       // round to nearest even
    unsigned short h = (unsigned short)(u >> 16);
    return __builtin_bit_cast(__bf16, h);
}

// Load a 16x32 bf16 WMMA operand (A layout; B uses same layout on transposed
// storage). Lane r = row, chunks at kbase+hi*8 and kbase+16+hi*8 (ISA 7.12.2).
static __device__ __forceinline__ v16bf ldop(const __bf16* base, int row, int ld,
                                             int kbase, int hi) {
    union { v16bf v; u32x4 q[2]; } u;
    const __bf16* p = base + (size_t)row * ld + kbase + hi * 8;
    u.q[0] = *(const u32x4*)p;
    u.q[1] = *(const u32x4*)(p + 16);
    return u.v;
}

static __device__ __forceinline__ float half_max16(float v) {
    #pragma unroll
    for (int m = 1; m < 16; m <<= 1) v = fmaxf(v, __shfl_xor(v, m, 32));
    return v;
}
static __device__ __forceinline__ float half_sum16(float v) {
    #pragma unroll
    for (int m = 1; m < 16; m <<= 1) v += __shfl_xor(v, m, 32);
    return v;
}

#define WMMA_BF16(A, B, C) \
    __builtin_amdgcn_wmma_f32_16x16x32_bf16(false, (A), false, (B), (short)0, (C), false, false)

// ---------------------------------------------------------------- converters
__global__ void k_cvt_hidden(const float* __restrict__ x, __bf16* __restrict__ y, int n) {
    for (int i = blockIdx.x * blockDim.x + threadIdx.x; i < n; i += gridDim.x * blockDim.x)
        y[i] = f2bf(x[i]);
}

// WT[mat][n][k] = W_mat[k][n]  (bf16 transpose of the three weight matrices)
__global__ void k_cvt_wT(const float* __restrict__ Wq, const float* __restrict__ Wk,
                         const float* __restrict__ Wv, __bf16* __restrict__ WT) {
    const int total = 3 * HH * HH;
    for (int t = blockIdx.x * blockDim.x + threadIdx.x; t < total; t += gridDim.x * blockDim.x) {
        int mat = t / (HH * HH);
        int rem = t % (HH * HH);
        int n = rem / HH, k = rem % HH;
        const float* W = (mat == 0) ? Wq : ((mat == 1) ? Wk : Wv);
        WT[t] = f2bf(W[k * HH + n]);
    }
}

// ------------------------------------------------------------- QKV projection
// One wave computes a 16x64 strip of one of Q/K/V = hidden @ W + b.
// Q,K stored bf16 [B,NH,S,HD]; V stored transposed bf16 [B,NH,HD,S].
__global__ void k_qkv(const __bf16* __restrict__ Hb, const __bf16* __restrict__ WT,
                      const float* __restrict__ bq, const float* __restrict__ bk,
                      const float* __restrict__ bv,
                      __bf16* __restrict__ Qb, __bf16* __restrict__ Kb,
                      __bf16* __restrict__ Vt) {
    const int wave = (blockIdx.x * blockDim.x + threadIdx.x) >> 5; // 0..9215
    const int lane = threadIdx.x & 31;
    const int r = lane & 15, hi = lane >> 4;

    const int mat    = wave / ((MM / 16) * (HH / 64));       // 0..2
    const int rem    = wave % ((MM / 16) * (HH / 64));
    const int mTile  = rem / (HH / 64);                      // 0..255
    const int nStrip = rem % (HH / 64);                      // 0..11

    const __bf16* Wt   = WT + (size_t)mat * HH * HH;
    const float*  bias = (mat == 0) ? bq : ((mat == 1) ? bk : bv);

    v8f c0 = {}, c1 = {}, c2 = {}, c3 = {};
    const int arow = mTile * 16 + r;
    for (int k = 0; k < HH; k += 32) {
        v16bf a = ldop(Hb, arow, HH, k, hi);
        v16bf b0 = ldop(Wt, nStrip * 64 +  0 + r, HH, k, hi);
        v16bf b1 = ldop(Wt, nStrip * 64 + 16 + r, HH, k, hi);
        v16bf b2 = ldop(Wt, nStrip * 64 + 32 + r, HH, k, hi);
        v16bf b3 = ldop(Wt, nStrip * 64 + 48 + r, HH, k, hi);
        c0 = WMMA_BF16(a, b0, c0);
        c1 = WMMA_BF16(a, b1, c1);
        c2 = WMMA_BF16(a, b2, c2);
        c3 = WMMA_BF16(a, b3, c3);
    }

    v8f acc[4] = {c0, c1, c2, c3};
    #pragma unroll
    for (int j = 0; j < 4; ++j) {
        const int n = nStrip * 64 + j * 16 + r;
        const float bval = bias[n];
        const int h = n >> 6, d = n & 63;
        #pragma unroll
        for (int i = 0; i < 8; ++i) {
            const int m = mTile * 16 + i + hi * 8;
            const int b = m >> 11, s = m & 2047;
            const __bf16 o = f2bf(acc[j][i] + bval);
            if (mat == 0)       Qb[(((size_t)(b * NHH + h) * SS + s) * HDD) + d] = o;
            else if (mat == 1)  Kb[(((size_t)(b * NHH + h) * SS + s) * HDD) + d] = o;
            else                Vt[(((size_t)(b * NHH + h) * HDD + d) * SS) + s] = o;
        }
    }
}

// ------------------------------------------------------------ flash attention
// One wave per (b, h, 16-query tile). Online softmax, 8 WMMAs / 32-key step.
__global__ void k_attn(const __bf16* __restrict__ Qb, const __bf16* __restrict__ Kb,
                       const __bf16* __restrict__ Vt, float* __restrict__ out) {
    __shared__ alignas(16) __bf16 plds[8][16 * 32];  // per-wave P tile (bf16)

    const int wid  = threadIdx.x >> 5;
    const int lane = threadIdx.x & 31;
    const int r = lane & 15, hi = lane >> 4;

    const int wave  = blockIdx.x * 8 + wid;          // 0..3071
    const int bh    = wave / (SS / 16);              // b*NH + h
    const int qTile = wave % (SS / 16);

    const __bf16* Qh = Qb + (size_t)bh * SS * HDD;
    const __bf16* Kh = Kb + (size_t)bh * SS * HDD;
    const __bf16* Vh = Vt + (size_t)bh * HDD * SS;

    const v16bf qa0 = ldop(Qh, qTile * 16 + r, HDD, 0,  hi);
    const v16bf qa1 = ldop(Qh, qTile * 16 + r, HDD, 32, hi);

    v8f o0 = {}, o1 = {}, o2 = {}, o3 = {};
    float mrow[8], lrow[8];
    #pragma unroll
    for (int i = 0; i < 8; ++i) { mrow[i] = -INFINITY; lrow[i] = 0.0f; }

    for (int kk = 0; kk < SS; kk += 32) {
        // S = Q * K^T  (two 16x16 tiles, K-dim = HD = 64)
        v16bf kb00 = ldop(Kh, kk +  0 + r, HDD, 0,  hi);
        v16bf kb01 = ldop(Kh, kk +  0 + r, HDD, 32, hi);
        v16bf kb10 = ldop(Kh, kk + 16 + r, HDD, 0,  hi);
        v16bf kb11 = ldop(Kh, kk + 16 + r, HDD, 32, hi);
        v8f z = {};
        v8f s0 = WMMA_BF16(qa0, kb00, z); s0 = WMMA_BF16(qa1, kb01, s0);
        v8f s1 = WMMA_BF16(qa0, kb10, z); s1 = WMMA_BF16(qa1, kb11, s1);

        // online softmax per row (row q = i + hi*8 lives in one half-wave)
        #pragma unroll
        for (int i = 0; i < 8; ++i) {
            float a0 = s0[i] * 0.125f;            // 1/sqrt(64)
            float a1 = s1[i] * 0.125f;
            float rmax = half_max16(fmaxf(a0, a1));
            float mnew = fmaxf(mrow[i], rmax);
            float corr = __expf(mrow[i] - mnew);
            float p0 = __expf(a0 - mnew);
            float p1 = __expf(a1 - mnew);
            float rsum = half_sum16(p0 + p1);
            lrow[i] = lrow[i] * corr + rsum;
            mrow[i] = mnew;
            o0[i] *= corr; o1[i] *= corr; o2[i] *= corr; o3[i] *= corr;
            // write P (D layout) to LDS as bf16
            plds[wid][(i + hi * 8) * 32 +      r] = f2bf(p0);
            plds[wid][(i + hi * 8) * 32 + 16 + r] = f2bf(p1);
        }
        asm volatile("s_wait_dscnt 0" ::: "memory");   // wave-local LDS RAW

        // re-load P in A-operand layout
        union { v16bf v; u32x4 q[2]; } up;
        up.q[0] = *(const u32x4*)&plds[wid][r * 32 +      hi * 8];
        up.q[1] = *(const u32x4*)&plds[wid][r * 32 + 16 + hi * 8];
        v16bf pa = up.v;

        // O += P * V  (V stored transposed: row d, contiguous in kk)
        v16bf vb0 = ldop(Vh,  0 + r, SS, kk, hi);
        v16bf vb1 = ldop(Vh, 16 + r, SS, kk, hi);
        v16bf vb2 = ldop(Vh, 32 + r, SS, kk, hi);
        v16bf vb3 = ldop(Vh, 48 + r, SS, kk, hi);
        o0 = WMMA_BF16(pa, vb0, o0);
        o1 = WMMA_BF16(pa, vb1, o1);
        o2 = WMMA_BF16(pa, vb2, o2);
        o3 = WMMA_BF16(pa, vb3, o3);
    }

    // epilogue: out[b, s, h*64 + d] = O / l
    const int b = bh / NHH, h = bh % NHH;
    #pragma unroll
    for (int i = 0; i < 8; ++i) {
        const int s = qTile * 16 + i + hi * 8;
        const float inv = 1.0f / lrow[i];
        float* orow = out + ((size_t)(b * SS + s)) * HH + h * HDD;
        orow[ 0 + r] = o0[i] * inv;
        orow[16 + r] = o1[i] * inv;
        orow[32 + r] = o2[i] * inv;
        orow[48 + r] = o3[i] * inv;
    }
}

// ---------------------------------------------------------------------- host
extern "C" void kernel_launch(void* const* d_in, const int* in_sizes, int n_in,
                              void* d_out, int out_size, void* d_ws, size_t ws_size,
                              hipStream_t stream) {
    const float* hs = (const float*)d_in[0];
    const float* Wq = (const float*)d_in[1];
    const float* bq = (const float*)d_in[2];
    const float* Wk = (const float*)d_in[3];
    const float* bk = (const float*)d_in[4];
    const float* Wv = (const float*)d_in[5];
    const float* bv = (const float*)d_in[6];
    float* out = (float*)d_out;

    // workspace layout (bf16 buffers)
    char* w = (char*)d_ws;
    const size_t szHb = (size_t)MM * HH * 2;            // 6.29 MB
    const size_t szWT = (size_t)3 * HH * HH * 2;        // 3.54 MB
    const size_t szQ  = (size_t)BB * NHH * SS * HDD * 2;// 6.29 MB each
    __bf16* Hb = (__bf16*)(w);
    __bf16* WT = (__bf16*)(w + szHb);
    __bf16* Qb = (__bf16*)(w + szHb + szWT);
    __bf16* Kb = (__bf16*)(w + szHb + szWT + szQ);
    __bf16* Vt = (__bf16*)(w + szHb + szWT + 2 * szQ);

    // 1) convert hidden + transpose/convert weights to bf16
    k_cvt_hidden<<<4096, 256, 0, stream>>>(hs, Hb, MM * HH);
    k_cvt_wT<<<4096, 256, 0, stream>>>(Wq, Wk, Wv, WT);

    // 2) QKV projections: 3 mats * 256 mTiles * 12 nStrips = 9216 waves
    k_qkv<<<1152, 256, 0, stream>>>(Hb, WT, bq, bk, bv, Qb, Kb, Vt);

    // 3) flash attention: 24 heads * 128 qTiles = 3072 waves
    k_attn<<<384, 256, 0, stream>>>(Qb, Kb, Vt, out);
}
